// MLPPredictor_34806414967174
// MI455X (gfx1250) — compile-verified, matrix operand order checked by reading, not run on hardware
//
#include <hip/hip_runtime.h>
#include <hip/hip_bf16.h>

typedef __attribute__((ext_vector_type(16))) _Float16 v16h;
typedef __attribute__((ext_vector_type(8)))  _Float16 v8h;
typedef __attribute__((ext_vector_type(8)))  float    v8f;

#define DIM      128
#define TWO_D    256
#define TILE_M   64
#define NTHREADS 256

// LDS layout (offsets/strides in _Float16 elements).
// Strides are multiples of 8 f16 (=16B) so v8h (b128) fragment loads stay aligned,
// and the +8 padding breaks the 64-bank stride pattern.
#define W1T_OFF 0          // [128 n][264]  (W1 transposed: w1t[n][k], k<256)
#define W1T_STR 264
#define W2T_OFF 33792      // [128 n][136]  (W2 transposed: w2t[n][k], k<128)
#define W2T_STR 136
#define WPT_OFF 51200      // [16  n][136]  (Wp transposed, n>=5 zero-padded)
#define WPT_STR 136
#define XH_OFF  53376      // [64  m][264]  gathered X (f16); reused for H2
#define XH_STR  264
#define HH_OFF  70272      // [64  m][136]  H1 (f16)
#define HH_STR  136
#define LDS_ELTS 78976     // 157,952 bytes total

static __device__ __forceinline__ v8f wmma32(v16h a, v16h b, v8f c) {
  // D = A(16x32 f16) * B(32x16 f16) + C(16x16 f32)
  return __builtin_amdgcn_wmma_f32_16x16x32_f16(
      /*neg_a=*/false, a, /*neg_b=*/false, b,
      /*c_mod=*/(short)0, c, /*reuse_a=*/false, /*reuse_b=*/false);
}

// Per-lane A/B fragment load per ISA "16-bit A-Matrix 16x32" layout:
// lane = M | (Khalf<<4); VGPRs 0..3 hold K = k0..k0+7 (2 f16/VGPR),
// VGPRs 4..7 hold K = k0+16..k0+23, where k0 = ks*32 + Khalf*8.
// Two 16-byte LDS reads (ds_load_b128) per fragment.
static __device__ __forceinline__ v16h load_frag(const _Float16* rowp, int k0) {
  v8h lo = *(const v8h*)(rowp + k0);
  v8h hi = *(const v8h*)(rowp + k0 + 16);
  v16h r;
#pragma unroll
  for (int i = 0; i < 8; ++i) { r[i] = lo[i]; r[i + 8] = hi[i]; }
  return r;
}

static __device__ __forceinline__ float gelu_exact(float x) {
  return 0.5f * x * (1.0f + erff(x * 0.70710678118654752440f));
}

__global__ __launch_bounds__(NTHREADS, 1)
void mlp_edge_predictor_kernel(const float* __restrict__ ufeat,
                               const float* __restrict__ ifeat,
                               const int*   __restrict__ src_idx,
                               const int*   __restrict__ dst_idx,
                               const float* __restrict__ W1,
                               const float* __restrict__ W2,
                               const float* __restrict__ Wp,
                               float* __restrict__ out,
                               int E) {
  __shared__ __align__(16) _Float16 lds[LDS_ELTS];

  const int tid  = threadIdx.x;
  const int lane = tid & 31;          // wave32
  const int wid  = tid >> 5;          // 8 waves
  const int m    = lane & 15;         // M (or N for B-fragments)
  const int kh   = lane >> 4;         // K-half select

  // ---- Load weights once per block, transposed + converted to f16 in LDS ----
  for (int i = tid; i < TWO_D * DIM; i += NTHREADS) {       // W1: [256][128] row-major
    int k = i >> 7, n = i & 127;
    lds[W1T_OFF + n * W1T_STR + k] = (_Float16)W1[i];
  }
  for (int i = tid; i < DIM * DIM; i += NTHREADS) {         // W2: [128][128]
    int k = i >> 7, n = i & 127;
    lds[W2T_OFF + n * W2T_STR + k] = (_Float16)W2[i];
  }
  for (int i = tid; i < 16 * DIM; i += NTHREADS) {          // Wp: [128][5] -> pad N to 16
    int k = i >> 4, n = i & 15;
    lds[WPT_OFF + n * WPT_STR + k] = (n < 5) ? (_Float16)Wp[k * 5 + n] : (_Float16)0.0f;
  }

  const int mt    = wid & 3;          // M-tile (0..3) of the 64-edge tile
  const int nbase = (wid >> 2) * 4;   // 4 N-tiles per wave (covers N=128 with 8 waves)
  const int ntiles = (E + TILE_M - 1) / TILE_M;

  for (int tile = blockIdx.x; tile < ntiles; tile += gridDim.x) {
    const int ebase = tile * TILE_M;
    __syncthreads();  // previous iteration's readers of XH are done; weights ready

    // ---- Gather: 64 edges x 256 f32 -> f16 in LDS (thread = column, coalesced) ----
    for (int r = 0; r < TILE_M; ++r) {
      int e = ebase + r;
      _Float16 v = (_Float16)0.0f;
      if (e < E) {
        int s = src_idx[e];
        int d = dst_idx[e];
        float f = (tid < DIM) ? ufeat[s * DIM + tid] : ifeat[d * DIM + (tid - DIM)];
        v = (_Float16)f;
      }
      lds[XH_OFF + r * XH_STR + tid] = v;
    }
    __syncthreads();

    // ---- GEMM1: H1 = gelu(X[64x256] @ W1[256x128]) ----
    {
      v8f acc[4] = {};
      const _Float16* arow = lds + XH_OFF + (mt * 16 + m) * XH_STR;
#pragma unroll
      for (int ks = 0; ks < 8; ++ks) {          // K = 256
        const int k0 = ks * 32 + kh * 8;
        v16h a = load_frag(arow, k0);
#pragma unroll
        for (int j = 0; j < 4; ++j) {
          const _Float16* brow = lds + W1T_OFF + ((nbase + j) * 16 + m) * W1T_STR;
          v16h b = load_frag(brow, k0);
          acc[j] = wmma32(a, b, acc[j]);
        }
      }
      // GELU epilogue in fp32, store f16 H1 (D-layout: row = v + kh*8, col = lane&15)
#pragma unroll
      for (int j = 0; j < 4; ++j) {
#pragma unroll
        for (int v = 0; v < 8; ++v) {
          int row = mt * 16 + v + kh * 8;
          int col = (nbase + j) * 16 + m;
          lds[HH_OFF + row * HH_STR + col] = (_Float16)gelu_exact(acc[j][v]);
        }
      }
    }
    __syncthreads();

    // ---- GEMM2: H2 = H1[64x128] @ W2[128x128]  (written into XH buffer) ----
    {
      v8f acc[4] = {};
      const _Float16* arow = lds + HH_OFF + (mt * 16 + m) * HH_STR;
#pragma unroll
      for (int ks = 0; ks < 4; ++ks) {          // K = 128
        const int k0 = ks * 32 + kh * 8;
        v16h a = load_frag(arow, k0);
#pragma unroll
        for (int j = 0; j < 4; ++j) {
          const _Float16* brow = lds + W2T_OFF + ((nbase + j) * 16 + m) * W2T_STR;
          v16h b = load_frag(brow, k0);
          acc[j] = wmma32(a, b, acc[j]);
        }
      }
#pragma unroll
      for (int j = 0; j < 4; ++j) {
#pragma unroll
        for (int v = 0; v < 8; ++v) {
          int row = mt * 16 + v + kh * 8;
          int col = (nbase + j) * 16 + m;
          lds[XH_OFF + row * XH_STR + col] = (_Float16)acc[j][v];
        }
      }
    }
    __syncthreads();

    // ---- GEMM3: S = H2[64x128] @ Wp[128x16(pad)] ; waves 0..3, one M-tile each ----
    if (wid < 4) {
      v8f acc = {};
      const _Float16* arow = lds + XH_OFF + (wid * 16 + m) * XH_STR;
      const _Float16* brow = lds + WPT_OFF + m * WPT_STR;   // n = lane&15
#pragma unroll
      for (int ks = 0; ks < 4; ++ks) {          // K = 128
        const int k0 = ks * 32 + kh * 8;
        v16h a = load_frag(arow, k0);
        v16h b = load_frag(brow, k0);
        acc = wmma32(a, b, acc);
      }
      // D-layout scatter to global: col (=lane&15) < 5 valid
      if (m < 5) {
#pragma unroll
        for (int v = 0; v < 8; ++v) {
          int row = wid * 16 + v + kh * 8;
          int e = ebase + row;
          if (e < E) out[e * 5 + m] = acc[v];
        }
      }
    }
    // top-of-loop __syncthreads() protects XH before the next gather
  }
}

extern "C" void kernel_launch(void* const* d_in, const int* in_sizes, int n_in,
                              void* d_out, int out_size, void* d_ws, size_t ws_size,
                              hipStream_t stream) {
  const float* ufeat   = (const float*)d_in[0];
  const float* ifeat   = (const float*)d_in[1];
  const int*   src_idx = (const int*)d_in[2];
  const int*   dst_idx = (const int*)d_in[3];
  const float* W1      = (const float*)d_in[4];
  const float* W2      = (const float*)d_in[5];
  const float* Wp      = (const float*)d_in[6];
  float*       out     = (float*)d_out;

  const int E = in_sizes[2];                       // number of edges
  const int ntiles = (E + TILE_M - 1) / TILE_M;    // 15625 for E=1M
  int grid = ntiles < 2048 ? ntiles : 2048;

  mlp_edge_predictor_kernel<<<grid, NTHREADS, 0, stream>>>(
      ufeat, ifeat, src_idx, dst_idx, W1, W2, Wp, out, E);
}
